// kspaceMap_12549894439467
// MI455X (gfx1250) — matrix operand order repaired and checked by reading, not compiled, not run on hardware
//
#include <hip/hip_runtime.h>

// Problem constants (from reference): input (32,2,128,128) f32, out (32,129,128,128) f32.
#define BZn   32
#define IMn   128
#define COUTn 129
#define TWO_PI 6.28318530717958647692f

typedef __attribute__((ext_vector_type(2)))  float  v2f;
typedef __attribute__((ext_vector_type(4)))  float  f4;
typedef __attribute__((ext_vector_type(8)))  float  v8f;
typedef __attribute__((ext_vector_type(16))) __bf16 v16bf;

// Pick f32 WMMA (16x16x4) if the builtin exists on device; else bf16 16x16x32.
// Host pass: always parse the f32 branch (amdgcn-builtin diagnostics are deferred).
#if !defined(__HIP_DEVICE_COMPILE__) || __has_builtin(__builtin_amdgcn_wmma_f32_16x16x4_f32)
#define USE_F32_WMMA 1
#else
#define USE_F32_WMMA 0
#endif

// ---------------------------------------------------------------------------
// Pass 1: DFT along W via WMMA.  X is (4096 x 128), rows = b*128+h.
//   t_re = X @ C   (C[v,k] = cos(2*pi*v*k/128))
//   t_im = -(X @ S)  (S[v,k] = sin(2*pi*v*k/128))
// One block = one 16-row tile of X (never crosses a batch boundary since 16|128).
// 8 waves per block cover the 8 k-tiles of 16 columns each.
// ---------------------------------------------------------------------------
__global__ __launch_bounds__(256) void kspace_dft_wmma(
    const float* __restrict__ in, float* __restrict__ t_re, float* __restrict__ t_im,
    float* __restrict__ cs_g, float* __restrict__ sn_g) {
  __shared__ __align__(16) float xs[16 * IMn];
  __shared__ float cs[IMn];
  __shared__ float sn[IMn];

  const int tid = threadIdx.x;
  if (tid < IMn) {
    const float ang = (TWO_PI / 128.0f) * (float)tid;
    const float c = cosf(ang), s = sinf(ang);
    cs[tid] = c; sn[tid] = s;
    if (blockIdx.x == 0) { cs_g[tid] = c; sn_g[tid] = s; }  // publish LUT for pass 2
  }

  // Stage 16 contiguous rows of x = input[:,0,:,:] into LDS (2048 floats, coalesced).
  const int r0 = blockIdx.x * 16;         // global row = b*128 + h
  const int b  = r0 >> 7;
  const int h0 = r0 & 127;
  const float* src = in + (size_t)b * (2 * IMn * IMn) + h0 * IMn;
  ((f4*)xs)[tid]       = ((const f4*)src)[tid];
  ((f4*)xs)[tid + 256] = ((const f4*)src)[tid + 256];
  __syncthreads();

  const int lane = tid & 31;
  const int wave = tid >> 5;
  const int N0   = wave * 16;             // k-tile start
  const int m    = lane & 15;
  const int half = lane >> 4;             // 0: lanes 0-15, 1: lanes 16-31
  const int kcol = N0 + m;                // this lane's B column (N = lane&15)

  v8f accC = {};                          // cos sums
  v8f accS = {};                          // sin sums

#if USE_F32_WMMA
  // A 16x4 f32 layout: lanes 0-15 -> {K0,K1}, lanes 16-31 -> {K2,K3}.
  // B 4x16 f32 layout: V0 -> rows K0(lanes 0-15)/K1(lanes 16-31), V1 -> K2/K3.
  for (int kk = 0; kk < 32; ++kk) {
    const int vA = kk * 4 + half * 2;
    v2f a;
    a.x = xs[m * IMn + vA];
    a.y = xs[m * IMn + vA + 1];
    const int vB = kk * 4 + half;
    v2f bc, bs;
    bc.x = cs[( vB      * kcol) & 127];
    bc.y = cs[((vB + 2) * kcol) & 127];
    bs.x = sn[( vB      * kcol) & 127];
    bs.y = sn[((vB + 2) * kcol) & 127];
    accC = __builtin_amdgcn_wmma_f32_16x16x4_f32(false, a, false, bc, (short)0, accC, false, false);
    accS = __builtin_amdgcn_wmma_f32_16x16x4_f32(false, a, false, bs, (short)0, accS, false, false);
  }
#else
  // bf16 fallback (wmma_f32_16x16x32_bf16), ISA 16-bit A 16x32 / B 32x16 layouts.
  for (int kk = 0; kk < 4; ++kk) {
    v16bf a, bc, bs;
#pragma unroll
    for (int e = 0; e < 16; ++e) {
      const int kA = kk * 32 + (e >> 3) * 16 + half * 8 + (e & 7);
      a[e] = (__bf16)xs[m * IMn + kA];
      const int kB = kk * 32 + (e >> 3) * 16 + half * 8 + (e & 7);
      bc[e] = (__bf16)cs[(kB * kcol) & 127];
      bs[e] = (__bf16)sn[(kB * kcol) & 127];
    }
    accC = __builtin_amdgcn_wmma_f32_16x16x32_bf16(false, a, false, bc, (short)0, accC, false, false);
    accS = __builtin_amdgcn_wmma_f32_16x16x32_bf16(false, a, false, bs, (short)0, accS, false, false);
  }
#endif

  // C/D layout: VGPR r -> M = half*8 + r, lane -> N = kcol.
  const int rowBase = r0 + half * 8;
#pragma unroll
  for (int r = 0; r < 8; ++r) {
    const int row = rowBase + r;
    t_re[row * IMn + kcol] =  accC[r];
    t_im[row * IMn + kcol] = -accS[r];
  }
}

// ---------------------------------------------------------------------------
// Pass 2: stream the 270 MB output.
//   c < 128: out[b,c,h,w] = (t_re[b,h,c]*cos(2pi*c*w/128) - t_im[b,h,c]*sin(...))/128
//   c == 128: copy input[b,1,h,w]
// grid = (16 h-chunks, 129 channels, 32 batches); wave = one (b,c,h) row,
// thread = 4 consecutive w -> float4 non-temporal store.
// ---------------------------------------------------------------------------
__global__ __launch_bounds__(256) void kspace_sep_write(
    const float* __restrict__ t_re, const float* __restrict__ t_im,
    const float* __restrict__ cs_g, const float* __restrict__ sn_g,
    const float* __restrict__ in, float* __restrict__ out) {
  __shared__ float cs[IMn];
  __shared__ float sn[IMn];
  if (threadIdx.x < IMn) {
    cs[threadIdx.x] = cs_g[threadIdx.x];
    sn[threadIdx.x] = sn_g[threadIdx.x];
  }
  __syncthreads();

  const int b = blockIdx.z;
  const int c = blockIdx.y;
  const int h = (blockIdx.x << 3) + (threadIdx.x >> 5);
  const int w = (threadIdx.x & 31) << 2;

  float* dst = out + ((size_t)(b * COUTn + c) * IMn + h) * IMn + w;

  if (c < IMn) {
    const int row = b * IMn + h;
    const float tre = t_re[row * IMn + c] * (1.0f / 128.0f);
    const float tim = t_im[row * IMn + c] * (1.0f / 128.0f);
    f4 v;
#pragma unroll
    for (int j = 0; j < 4; ++j) {
      const int idx = (c * (w + j)) & 127;   // exact angle mod 2*pi
      v[j] = tre * cs[idx] - tim * sn[idx];
    }
    __builtin_nontemporal_store(v, (f4*)dst);
  } else {
    const float* srcp = in + (size_t)b * (2 * IMn * IMn) + IMn * IMn + h * IMn + w;
    f4 v = *(const f4*)srcp;
    __builtin_nontemporal_store(v, (f4*)dst);
  }
}

// ---------------------------------------------------------------------------
extern "C" void kernel_launch(void* const* d_in, const int* in_sizes, int n_in,
                              void* d_out, int out_size, void* d_ws, size_t ws_size,
                              hipStream_t stream) {
  const float* input = (const float*)d_in[0];   // (32,2,128,128) f32; d_in[1] (mask) unused by reference
  float* t_re = (float*)d_ws;                    // 4096*128 floats
  float* t_im = t_re + BZn * IMn * IMn;          // 4096*128 floats
  float* cs_g = t_im + BZn * IMn * IMn;          // 128 floats
  float* sn_g = cs_g + IMn;                      // 128 floats

  kspace_dft_wmma<<<dim3(256), dim3(256), 0, stream>>>(input, t_re, t_im, cs_g, sn_g);
  kspace_sep_write<<<dim3(IMn / 8, COUTn, BZn), dim3(256), 0, stream>>>(
      t_re, t_im, cs_g, sn_g, input, (float*)d_out);
}
